// JointNetwork_30039001269068
// MI455X (gfx1250) — matrix operand order repaired and checked by reading, not compiled, not run on hardware
//
#include <hip/hip_runtime.h>
#include <stdint.h>

typedef __attribute__((ext_vector_type(2)))  float    v2f;
typedef __attribute__((ext_vector_type(8)))  float    v8f;
typedef __attribute__((ext_vector_type(8)))  _Float16 v8h;
typedef __attribute__((ext_vector_type(16))) _Float16 v16h;

constexpr int kD        = 1024;
constexpr int kVocab    = 5001;
constexpr int kNPad     = 5120;          // N padded to multiple of 256
constexpr int kNT       = kNPad / 16;    // 320 column tiles of 16
constexpr int kB        = 4;
constexpr int kT        = 200;
constexpr int kU        = 50;
constexpr int kMTot     = kB * kT * kU;  // 40000 = 625 * 64
constexpr int kEncRows  = kB * kT;       // 800
constexpr int kPredRows = kB * kU;       // 200
constexpr int kTileNPerBlock = kNPad / 256;  // 20
constexpr int kSteps    = kTileNPerBlock * 32; // 640

// ---------------------------------------------------------------------------
// Repack W_out (f32, [1024][5001]) into f16 fragment-major layout:
//   Bp[ kb ][ nTile ][ lane ][ 16 halves ]   (WMMA-B swizzle, zero padded)
// ---------------------------------------------------------------------------
__global__ void repack_wout(const float* __restrict__ Wout,
                            _Float16* __restrict__ Bp) {
    int tid = blockIdx.x * blockDim.x + threadIdx.x;   // one thread per 16-half slot
    if (tid >= 32 * kNT * 32) return;
    int lane  = tid & 31;
    int nTile = (tid >> 5) % kNT;
    int kb    = tid / (kNT * 32);
    int n     = nTile * 16 + (lane & 15);
    int kOff  = kb * 32 + ((lane & 16) ? 16 : 0);
    v8h lo, hi;
#pragma unroll
    for (int j = 0; j < 8; ++j)
        lo[j] = (_Float16)((n < kVocab) ? Wout[(kOff + j) * kVocab + n] : 0.0f);
#pragma unroll
    for (int j = 0; j < 8; ++j)
        hi[j] = (_Float16)((n < kVocab) ? Wout[(kOff + 8 + j) * kVocab + n] : 0.0f);
    v8h* dst = (v8h*)(Bp + (size_t)tid * 16);
    dst[0] = lo;
    dst[1] = hi;
}

// ---------------------------------------------------------------------------
// Projections in full f32 via V_WMMA_F32_16X16X4_F32; one wave per 16x16 tile.
// ---------------------------------------------------------------------------
__global__ __launch_bounds__(256)
void proj_kernel(const float* __restrict__ enc,  const float* __restrict__ pred,
                 const float* __restrict__ We,   const float* __restrict__ be,
                 const float* __restrict__ Wp,   const float* __restrict__ bp,
                 float* __restrict__ encP,       float* __restrict__ predP) {
    const int lane  = threadIdx.x & 31;
    const int wave  = threadIdx.x >> 5;
    const int mTile = blockIdx.x;                 // 0..62  (50 enc + 13 pred)
    const int nTile = blockIdx.y * 8 + wave;      // 0..63

    const float* X;  const float* W;  const float* bias;  float* Y;
    int rows, mBase;
    if (mTile < 50) { X = enc;  W = We; bias = be; Y = encP;  rows = kEncRows;  mBase = mTile * 16; }
    else            { X = pred; W = Wp; bias = bp; Y = predP; rows = kPredRows; mBase = (mTile - 50) * 16; }

    const int ml   = lane & 15;
    const int m    = min(mBase + ml, rows - 1);
    const int koff = (lane & 16) ? 2 : 0;
    const int n    = nTile * 16 + ml;
    const float* xrow = X + m * kD;

    v8f c = {};
#pragma unroll 4
    for (int k = 0; k < kD; k += 4) {
        const float2 av = *(const float2*)(xrow + k + koff);
        v2f a; a[0] = av.x; a[1] = av.y;
        v2f b;
        b[0] = W[(k + koff) * kD + n];
        b[1] = W[(k + koff + 1) * kD + n];
        c = __builtin_amdgcn_wmma_f32_16x16x4_f32(false, a, false, b,
                                                  (short)0, c, false, false);
    }

    const float bv  = bias[n];
    const int mrow0 = mBase + ((lane & 16) ? 8 : 0);
    if (mBase + 16 <= rows) {            // full-tile fast path, no guards
#pragma unroll
        for (int v = 0; v < 8; ++v) Y[(mrow0 + v) * kD + n] = c[v] + bv;
    } else {
#pragma unroll
        for (int v = 0; v < 8; ++v) {
            int mo = mrow0 + v;
            if (mo < rows) Y[mo * kD + n] = c[v] + bv;
        }
    }
}

// ---------------------------------------------------------------------------
// Main joint GEMM.
//   Block = 128 threads (4 waves), tileM = 64 rows (625 blocks, exact).
//   A panel: full-K joint tile relu(enc_p+pred_p) -> f16 in LDS once (128 KB).
//   B tiles: 32(K) x 256(N) f16, double-buffered in LDS via
//            global_load_async_to_lds_b128 (ASYNCcnt), one barrier per K-step.
//   Wave register tile: 64(M) x 64(N) = 16 C fragments, f32 accumulate.
// ---------------------------------------------------------------------------
__global__ __launch_bounds__(128)
void joint_gemm(const float* __restrict__ encP, const float* __restrict__ predP,
                const _Float16* __restrict__ Bp, const float* __restrict__ bout,
                float* __restrict__ out) {
    __shared__ __align__(32) _Float16 smemA[32 * 4 * 32 * 16];   // 65536 halves = 128 KB
    __shared__ __align__(32) _Float16 smemB[2][16 * 32 * 16];    // 2 x 8192 halves = 32 KB

    const int tid   = threadIdx.x;
    const int lane  = tid & 31;
    const int wv    = tid >> 5;       // waveN: 0..3
    const int tileM = blockIdx.x;     // 0..624

    // Issue async copy of B tile for linear step s into LDS buffer `buf`.
    auto issueB = [&](int s, int buf) {
        const int tN = s >> 5;
        const int kb = s & 31;
#pragma unroll
        for (int q = 0; q < 4; ++q) {
            const int slot = q * 128 + tid;            // 0..511 (16 nTiles x 32 lanes)
            const int nLoc = slot >> 5;
            const int ln   = slot & 31;
            const size_t srcHalf =
                ((size_t)(kb * kNT + tN * 16 + nLoc) * 32 + ln) * 16;
            const uint64_t g0 = (uint64_t)(uintptr_t)(Bp + srcHalf);
            const uint32_t l0 = (uint32_t)(uintptr_t)(&smemB[buf][slot * 16]);
            asm volatile("global_load_async_to_lds_b128 %0, %1, off"
                         :: "v"(l0), "v"(g0) : "memory");
            asm volatile("global_load_async_to_lds_b128 %0, %1, off offset:16"
                         :: "v"(l0), "v"(g0) : "memory");
        }
    };

    // Prologue: start step-0 B copy, then build the A panel.
    issueB(0, 0);
    {
        const int mt = wv;                                     // 0..3
        const int m  = tileM * 64 + mt * 16 + (lane & 15);     // always < 40000
        const int bb = m / (kT * kU);
        const int rem = m % (kT * kU);
        const int t = rem / kU;
        const int u = rem % kU;
        const float* erow = encP  + (size_t)(bb * kT + t) * kD;
        const float* prow = predP + (size_t)(bb * kU + u) * kD;
        const int khalf = (lane & 16) ? 16 : 0;
        for (int kb = 0; kb < 32; ++kb) {
            const int k0 = kb * 32 + khalf;
            float4 ea = *(const float4*)(erow + k0);
            float4 eb = *(const float4*)(erow + k0 + 4);
            float4 ec = *(const float4*)(erow + k0 + 8);
            float4 ed = *(const float4*)(erow + k0 + 12);
            float4 pa = *(const float4*)(prow + k0);
            float4 pb = *(const float4*)(prow + k0 + 4);
            float4 pc = *(const float4*)(prow + k0 + 8);
            float4 pd = *(const float4*)(prow + k0 + 12);
            v8h lo, hi;
            lo[0] = (_Float16)fmaxf(ea.x + pa.x, 0.0f);
            lo[1] = (_Float16)fmaxf(ea.y + pa.y, 0.0f);
            lo[2] = (_Float16)fmaxf(ea.z + pa.z, 0.0f);
            lo[3] = (_Float16)fmaxf(ea.w + pa.w, 0.0f);
            lo[4] = (_Float16)fmaxf(eb.x + pb.x, 0.0f);
            lo[5] = (_Float16)fmaxf(eb.y + pb.y, 0.0f);
            lo[6] = (_Float16)fmaxf(eb.z + pb.z, 0.0f);
            lo[7] = (_Float16)fmaxf(eb.w + pb.w, 0.0f);
            hi[0] = (_Float16)fmaxf(ec.x + pc.x, 0.0f);
            hi[1] = (_Float16)fmaxf(ec.y + pc.y, 0.0f);
            hi[2] = (_Float16)fmaxf(ec.z + pc.z, 0.0f);
            hi[3] = (_Float16)fmaxf(ec.w + pc.w, 0.0f);
            hi[4] = (_Float16)fmaxf(ed.x + pd.x, 0.0f);
            hi[5] = (_Float16)fmaxf(ed.y + pd.y, 0.0f);
            hi[6] = (_Float16)fmaxf(ed.z + pd.z, 0.0f);
            hi[7] = (_Float16)fmaxf(ed.w + pd.w, 0.0f);
            v8h* dst = (v8h*)(smemA + ((size_t)(kb * 4 + mt) * 32 + lane) * 16);
            dst[0] = lo;
            dst[1] = hi;
        }
    }

    v8f c[4][4];
    for (int tN = 0; tN < kTileNPerBlock; ++tN) {
#pragma unroll
        for (int mi = 0; mi < 4; ++mi)
#pragma unroll
            for (int nf = 0; nf < 4; ++nf) c[mi][nf] = (v8f){};

        for (int kb = 0; kb < 32; ++kb) {
            const int s = tN * 32 + kb;
            // My buf[s&1] copies (issued last step) must be in LDS, then sync all.
            asm volatile("s_wait_asynccnt 0x0" ::: "memory");
            __syncthreads();
            // Prefetch next step's B tile (safe: everyone is past last read of it).
            if (s + 1 < kSteps) issueB(s + 1, (s + 1) & 1);

            const int cur = s & 1;
            v16h a0 = *(const v16h*)(smemA + ((size_t)(kb * 4 + 0) * 32 + lane) * 16);
            v16h a1 = *(const v16h*)(smemA + ((size_t)(kb * 4 + 1) * 32 + lane) * 16);
            v16h a2 = *(const v16h*)(smemA + ((size_t)(kb * 4 + 2) * 32 + lane) * 16);
            v16h a3 = *(const v16h*)(smemA + ((size_t)(kb * 4 + 3) * 32 + lane) * 16);
#pragma unroll
            for (int nf = 0; nf < 4; ++nf) {
                v16h b = *(const v16h*)(&smemB[cur][((wv * 4 + nf) * 32 + lane) * 16]);
                c[0][nf] = __builtin_amdgcn_wmma_f32_16x16x32_f16(false, a0, false, b,
                                                                  (short)0, c[0][nf], false, false);
                c[1][nf] = __builtin_amdgcn_wmma_f32_16x16x32_f16(false, a1, false, b,
                                                                  (short)0, c[1][nf], false, false);
                c[2][nf] = __builtin_amdgcn_wmma_f32_16x16x32_f16(false, a2, false, b,
                                                                  (short)0, c[2][nf], false, false);
                c[3][nf] = __builtin_amdgcn_wmma_f32_16x16x32_f16(false, a3, false, b,
                                                                  (short)0, c[3][nf], false, false);
            }
        }

        // Store this N-stripe with fused b_out. Rows always valid (625*64 = 40000).
        const int nl   = lane & 15;
        const int mAdd = (lane & 16) ? 8 : 0;
#pragma unroll
        for (int nf = 0; nf < 4; ++nf) {
            const int n = tN * 256 + wv * 64 + nf * 16 + nl;
            if (n < kVocab) {
                const float bv = bout[n];
#pragma unroll
                for (int mi = 0; mi < 4; ++mi) {
                    const int m0 = tileM * 64 + mi * 16 + mAdd;
#pragma unroll
                    for (int v = 0; v < 8; ++v)
                        out[(size_t)(m0 + v) * kVocab + n] = c[mi][nf][v] + bv;
                }
            }
        }
    }
}

// ---------------------------------------------------------------------------
extern "C" void kernel_launch(void* const* d_in, const int* in_sizes, int n_in,
                              void* d_out, int out_size, void* d_ws, size_t ws_size,
                              hipStream_t stream) {
    const float* enc  = (const float*)d_in[0];
    const float* pred = (const float*)d_in[1];
    const float* We   = (const float*)d_in[2];
    const float* be   = (const float*)d_in[3];
    const float* Wp   = (const float*)d_in[4];
    const float* bp   = (const float*)d_in[5];
    const float* Wout = (const float*)d_in[6];
    const float* bout = (const float*)d_in[7];
    float* out = (float*)d_out;

    // Workspace: enc_p (f32) | pred_p (f32) | Bpack (f16)  ~14.6 MB
    float*    encP  = (float*)d_ws;
    float*    predP = encP + (size_t)kEncRows * kD;
    _Float16* Bpack = (_Float16*)(predP + (size_t)kPredRows * kD);

    const int repackThreads = 32 * kNT * 32;
    hipLaunchKernelGGL(repack_wout, dim3((repackThreads + 255) / 256), dim3(256),
                       0, stream, Wout, Bpack);
    hipLaunchKernelGGL(proj_kernel, dim3(63, 8), dim3(256), 0, stream,
                       enc, pred, We, be, Wp, bp, encP, predP);
    hipLaunchKernelGGL(joint_gemm, dim3(625), dim3(128), 0, stream,
                       encP, predP, Bpack, bout, out);
}
